// near_refine_78752520339659
// MI455X (gfx1250) — matrix operand order repaired and checked by reading, not compiled, not run on hardware
//
#include <hip/hip_runtime.h>
#include <math.h>

// ---------------------------------------------------------------------------
// Problem constants (reference: B,C,H,W = 4,256,56,56 ; N = 3136 = 49*64)
// ---------------------------------------------------------------------------
constexpr int Bb = 4;
constexpr int Cc = 256;
constexpr int Nn = 3136;
constexpr float SCALE = 0.0625f;   // 1/sqrt(256)

typedef __attribute__((ext_vector_type(16))) __bf16 v16bf;
typedef __attribute__((ext_vector_type(8)))  __bf16 v8bf;
typedef __attribute__((ext_vector_type(8)))  float  v8f;
typedef __attribute__((ext_vector_type(4)))  unsigned int v4u;
typedef __attribute__((ext_vector_type(8)))  int v8i;
typedef __attribute__((ext_vector_type(4)))  int v4i;

#define DEVI __device__ __forceinline__

#if __has_builtin(__builtin_amdgcn_tensor_load_to_lds) && \
    __has_builtin(__builtin_amdgcn_s_wait_tensorcnt)
#define HAVE_TDM 1
#else
#define HAVE_TDM 0
#endif

DEVI v16bf frag_cat(v8bf lo, v8bf hi) {
  return __builtin_shufflevector(lo, hi, 0,1,2,3,4,5,6,7,8,9,10,11,12,13,14,15);
}

// A fragment: 16(M) x 32(K) bf16 from row-major [m][k] (ld elements).
// lane L: m = L&15, half = L>>4 ; elems 0..7 -> K = 8*half+e ; 8..15 -> K = 16+8*half+(e-8)
DEVI v16bf load_frag_a(const __bf16* src, int ld) {
  const int lane = threadIdx.x & 31;
  const __bf16* p = src + (size_t)(lane & 15) * ld + 8 * (lane >> 4);
  v8bf lo = *(const v8bf*)p;
  v8bf hi = *(const v8bf*)(p + 16);
  return frag_cat(lo, hi);
}

// B fragment: 32(K) x 16(N) bf16 with source stored [n][k] (k contiguous, ld elements).
// lane L: n = L&15, half = L>>4 ; elems 0..7 -> K = 8*half+e ; 8..15 -> K = 16+8*half+(e-8)
DEVI v16bf load_frag_b(const __bf16* src, int ld) {
  const int lane = threadIdx.x & 31;
  const __bf16* p = src + (size_t)(lane & 15) * ld + 8 * (lane >> 4);
  v8bf lo = *(const v8bf*)p;
  v8bf hi = *(const v8bf*)(p + 16);
  return frag_cat(lo, hi);
}

DEVI v8f wmma_bf16(v16bf a, v16bf b, v8f c) {
  return __builtin_amdgcn_wmma_f32_16x16x32_bf16(false, a, false, b, (short)0, c,
                                                 false, false);
}

#if HAVE_TDM
// Tensor Data Mover: load a 2D tile (rows x row_e8, in 8-byte elements) from
// global memory (row stride gstride_e8) into LDS (packed). D# per ISA 08 §8.
// Wave-level op (EXEC ignored); caller restricts issue to one wave.
// This toolchain's builtin takes 6 args: (g0, g1, g2, g3, extra_g, cpol).
DEVI void tdm_load(const void* gsrc, void* ldst, int rows, int row_e8,
                   int gstride_e8) {
  unsigned long long ga = (unsigned long long)(size_t)gsrc;
  unsigned lab = (unsigned)(size_t)ldst;          // LDS byte address
  v4u g0;
  g0[0] = 1u;                                     // count=1, user D#
  g0[1] = lab;                                    // lds_addr
  g0[2] = (unsigned)(ga & 0xffffffffu);           // global_addr[31:0]
  g0[3] = (unsigned)((ga >> 32) & 0x01ffffffu) |  // global_addr[56:32]
          (2u << 30);                              // type = 2 ("image")
  const unsigned td0 = (unsigned)row_e8;          // tensor_dim0 == tile row
  const unsigned td1 = (unsigned)rows;            // tensor_dim1 == #rows
  v8i g1;
  g1[0] = (int)(3u << 16);                        // data_size=3 (8B units)
  g1[1] = (int)((td0 & 0xffffu) << 16);           // tensor_dim0[15:0]
  g1[2] = (int)(((td0 >> 16) & 0xffffu) | ((td1 & 0xffffu) << 16));
  g1[3] = (int)(((td1 >> 16) & 0xffffu) |
                (((unsigned)row_e8 & 0xffffu) << 16));   // tile_dim0
  g1[4] = (int)((unsigned)rows & 0xffffu);        // tile_dim1 ; tile_dim2=0
  g1[5] = (int)(unsigned)gstride_e8;              // tensor_dim0_stride[31:0]
  g1[6] = 0;                                      // stride0 hi, stride1 lo
  g1[7] = 0;                                      // stride1 hi
  v4i g2; g2[0] = 1; g2[1] = 1; g2[2] = 0; g2[3] = 0;  // dim2=dim3=1 (benign)
  v4i g3; g3[0] = 0; g3[1] = (int)(1u << 16); g3[2] = 0; g3[3] = 0; // dim4=1
  v8i gx; gx[0]=0; gx[1]=0; gx[2]=0; gx[3]=0; gx[4]=0; gx[5]=0; gx[6]=0; gx[7]=0;
  __builtin_amdgcn_tensor_load_to_lds(g0, g1, g2, g3, gx, 0);
}
#endif

// ---------------------------------------------------------------------------
// Prep kernels
// ---------------------------------------------------------------------------
__global__ void conv_weights_kernel(const float* __restrict__ Wv,
                                    const float* __restrict__ Wl,
                                    __bf16* __restrict__ Wvb,
                                    __bf16* __restrict__ Wl1,
                                    __bf16* __restrict__ Wl2) {
  int i = blockIdx.x * 256 + threadIdx.x;
  if (i < Cc * Cc) {
    int m = i / Cc, k = i % Cc;
    Wvb[i] = (__bf16)Wv[i];
    Wl1[i] = (__bf16)Wl[m * 2 * Cc + k];
    Wl2[i] = (__bf16)Wl[m * 2 * Cc + Cc + k];
  }
}

__global__ void copy_feat0_kernel(const float* __restrict__ f0,
                                  float* __restrict__ out) {
  size_t i = (size_t)blockIdx.x * 256 + threadIdx.x;
  size_t per_b = (size_t)Cc * Nn;
  if (i < (size_t)Bb * per_b) {
    size_t b = i / per_b, r = i % per_b;
    out[(b * 4) * per_b + r] = f0[i];
  }
}

// feat [b][C][N] f32 -> tok [b][N][C] bf16 (LDS-tiled transpose)
__global__ void transpose_tok_kernel(const float* __restrict__ feat,
                                     __bf16* __restrict__ tok) {
  __shared__ float tile[32][33];
  const int b = blockIdx.z;
  const int c0 = blockIdx.y * 32;
  const int n0 = blockIdx.x * 32;
  const int col = threadIdx.x & 31;
  const int row = threadIdx.x >> 5;                 // 0..7
  const float* fb = feat + (size_t)b * Cc * Nn;
  __bf16* tb = tok + (size_t)b * Nn * Cc;
#pragma unroll
  for (int i = 0; i < 4; ++i)
    tile[row + 8 * i][col] = fb[(size_t)(c0 + row + 8 * i) * Nn + n0 + col];
  __syncthreads();
#pragma unroll
  for (int i = 0; i < 4; ++i)
    tb[(size_t)(n0 + row + 8 * i) * Cc + c0 + col] =
        (__bf16)tile[col][row + 8 * i];
}

// ---------------------------------------------------------------------------
// V-linear: vT[b][m][n] = sum_k Wv[m][k] * ftok[b][n][k] + bv[m]   (bf16 out)
// grid (49, 4, B), block 128. Software-pipelined: all 8 A frags preloaded,
// B frags double-buffered so loads for kc+1 overlap WMMAs for kc.
// ---------------------------------------------------------------------------
__global__ void gemm_vlinear_kernel(const __bf16* __restrict__ Wvb,
                                    const __bf16* __restrict__ ftok,
                                    const float* __restrict__ bv,
                                    __bf16* __restrict__ vT) {
  const int b = blockIdx.z;
  const int wave = threadIdx.x >> 5, lane = threadIdx.x & 31;
  const int lc = lane & 15, lh = lane >> 4;
  const int m0 = blockIdx.y * 64 + wave * 16;
  const int n0 = blockIdx.x * 64;
  const __bf16* fb = ftok + (size_t)b * Nn * Cc;
  __bf16* vb = vT + (size_t)b * Cc * Nn;

  v8f acc[4];
#pragma unroll
  for (int t = 0; t < 4; ++t)
#pragma unroll
    for (int r = 0; r < 8; ++r) acc[t][r] = bv[m0 + r + 8 * lh];

  v16bf afr[8];
#pragma unroll
  for (int kc = 0; kc < 8; ++kc)
    afr[kc] = load_frag_a(Wvb + (size_t)m0 * Cc + kc * 32, Cc);

  v16bf bb0[4], bb1[4];
#pragma unroll
  for (int t = 0; t < 4; ++t)
    bb0[t] = load_frag_b(fb + (size_t)(n0 + t * 16) * Cc, Cc);

#pragma unroll
  for (int kc = 0; kc < 8; ++kc) {
    v16bf* cur = (kc & 1) ? bb1 : bb0;
    v16bf* nxt = (kc & 1) ? bb0 : bb1;
    if (kc < 7) {
#pragma unroll
      for (int t = 0; t < 4; ++t)
        nxt[t] = load_frag_b(fb + (size_t)(n0 + t * 16) * Cc + (kc + 1) * 32, Cc);
    }
#pragma unroll
    for (int t = 0; t < 4; ++t) acc[t] = wmma_bf16(afr[kc], cur[t], acc[t]);
  }
#pragma unroll
  for (int t = 0; t < 4; ++t)
#pragma unroll
    for (int r = 0; r < 8; ++r)
      vb[(size_t)(m0 + r + 8 * lh) * Nn + n0 + t * 16 + lc] = (__bf16)acc[t][r];
}

// ---------------------------------------------------------------------------
// Flash attention stage:
//   refined[b][n][c] = softmax_m(q·f^T * scale + mask)[n][m] · v[m][c] + f[n][c]
// grid (49, B), block 128 (4 waves; wave owns 16 query rows; 64 rows / block)
// LDS: Q 32K + 2x F 32K + 2x Vt 32K + P 8K = 168 KB.  Tiles staged by the
// Tensor Data Mover (TENSOR_LOAD_TO_LDS, wave 0 issues, s_wait_tensorcnt),
// double-buffered so tile kt+1 streams in while tile kt is computed.
// ---------------------------------------------------------------------------
__global__ void attn_kernel(const __bf16* __restrict__ qtok,   // [B][N][C]
                            const __bf16* __restrict__ ftok,   // [B][N][C]
                            const __bf16* __restrict__ vT,     // [B][C][N]
                            const float* __restrict__ mask,    // [N][N]
                            const float* __restrict__ featf,   // [B][C][N]
                            __bf16* __restrict__ rtok) {       // [B][N][C]
  extern __shared__ char smem[];
  __bf16* Qs  = (__bf16*)smem;            // 64 x 256
  __bf16* Fs[2]; Fs[0] = Qs + 64 * Cc; Fs[1] = Fs[0] + 64 * Cc;
  __bf16* Vs[2]; Vs[0] = Fs[1] + 64 * Cc; Vs[1] = Vs[0] + Cc * 64;
  __bf16* Ps = Vs[1] + Cc * 64;           // 4 waves x (16 x 64)

  const int b = blockIdx.y;
  const int row0 = blockIdx.x * 64;
  const int tid = threadIdx.x;
  const int wave = tid >> 5, lane = tid & 31;
  const int lc = lane & 15, lh = lane >> 4;

  const __bf16* qb = qtok + (size_t)b * Nn * Cc;
  const __bf16* fb = ftok + (size_t)b * Nn * Cc;
  const __bf16* vb = vT + (size_t)b * Cc * Nn;
  const float* rb = featf + (size_t)b * Cc * Nn;
  __bf16* ob = rtok + (size_t)b * Nn * Cc;

  // Cooperative tile staging: F tile [64][256] contiguous, Vt tile [256][64]
  // strided (row stride N bf16 = 784 x 8B).
  auto stage_tiles = [&](int buf, int key0) {
#if HAVE_TDM
    if (wave == 0) {
      tdm_load(fb + (size_t)key0 * Cc, Fs[buf], 1, 64 * Cc / 4, 64 * Cc / 4);
      tdm_load(vb + key0, Vs[buf], Cc, 64 / 4, Nn / 4);
    }
#else
    const __bf16* fsrc = fb + (size_t)key0 * Cc;
    for (int i = tid; i < 64 * Cc / 8; i += 128)
      *(v8bf*)(Fs[buf] + (size_t)i * 8) = *(const v8bf*)(fsrc + (size_t)i * 8);
    for (int i = tid; i < Cc * 64 / 8; i += 128) {
      int c = i >> 3, chunk = i & 7;
      *(v8bf*)(Vs[buf] + (size_t)i * 8) =
          *(const v8bf*)(vb + (size_t)c * Nn + key0 + chunk * 8);
    }
#endif
  };
  auto wait_tiles = [&]() {
#if HAVE_TDM
    if (wave == 0) __builtin_amdgcn_s_wait_tensorcnt(0);
#endif
    __syncthreads();
  };

  // Stage the 64x256 query block (once) and the first K/V tile.
#if HAVE_TDM
  if (wave == 0) tdm_load(qb + (size_t)row0 * Cc, Qs, 1, 64 * Cc / 4, 64 * Cc / 4);
#else
  {
    const __bf16* src = qb + (size_t)row0 * Cc;
    for (int i = tid; i < 64 * Cc / 8; i += 128)
      *(v8bf*)(Qs + (size_t)i * 8) = *(const v8bf*)(src + (size_t)i * 8);
  }
#endif
  stage_tiles(0, 0);
  wait_tiles();

  v8f O[16];
#pragma unroll
  for (int ct = 0; ct < 16; ++ct)
#pragma unroll
    for (int r = 0; r < 8; ++r) O[ct][r] = 0.0f;

  float rm[8], rl[8];
#pragma unroll
  for (int r = 0; r < 8; ++r) { rm[r] = -1e30f; rl[r] = 0.0f; }

  const int my_r0 = row0 + wave * 16 + 8 * lh;   // first of this lane's 8 rows
  __bf16* Pw = Ps + wave * 16 * 64;
  constexpr int NT = Nn / 64;                    // 49 key tiles

  for (int kt = 0; kt < NT; ++kt) {
    const int key0 = kt * 64;
    const __bf16* Fc = Fs[kt & 1];
    const __bf16* Vc = Vs[kt & 1];

    // Kick off TDM staging of the next tile while computing this one.
    if (kt + 1 < NT) {
      stage_tiles((kt + 1) & 1, key0 + 64);
      __builtin_prefetch(fb + (size_t)(key0 + 128) * Cc, 0, 1);
    }

    // S(16x64) = Q_tile x F_tile^T
    v8f S[4];
#pragma unroll
    for (int t = 0; t < 4; ++t)
#pragma unroll
      for (int r = 0; r < 8; ++r) S[t][r] = 0.0f;
#pragma unroll
    for (int kc = 0; kc < 8; ++kc) {
      v16bf a = load_frag_a(Qs + (size_t)wave * 16 * Cc + kc * 32, Cc);
#pragma unroll
      for (int t = 0; t < 4; ++t) {
        v16bf bf = load_frag_b(Fc + (size_t)t * 16 * Cc + kc * 32, Cc);
        S[t] = wmma_bf16(a, bf, S[t]);
      }
    }

    // scale + mask; per-row max over this 64-key tile
    float cmax[8];
#pragma unroll
    for (int r = 0; r < 8; ++r) cmax[r] = -1e30f;
#pragma unroll
    for (int t = 0; t < 4; ++t)
#pragma unroll
      for (int r = 0; r < 8; ++r) {
        float x = S[t][r] * SCALE +
                  mask[(size_t)(my_r0 + r) * Nn + key0 + t * 16 + lc];
        S[t][r] = x;
        cmax[r] = fmaxf(cmax[r], x);
      }
#pragma unroll
    for (int r = 0; r < 8; ++r)
#pragma unroll
      for (int m = 1; m < 16; m <<= 1)
        cmax[r] = fmaxf(cmax[r], __shfl_xor(cmax[r], m, 16));

    float fac[8], rsum[8];
#pragma unroll
    for (int r = 0; r < 8; ++r) {
      float nm = fmaxf(rm[r], cmax[r]);
      fac[r] = __expf(rm[r] - nm);
      rm[r] = nm;
      rsum[r] = 0.0f;
    }

    // P = exp(S - m); stash per-wave into LDS in [row][key] layout
#pragma unroll
    for (int t = 0; t < 4; ++t)
#pragma unroll
      for (int r = 0; r < 8; ++r) {
        float p = __expf(S[t][r] - rm[r]);
        rsum[r] += p;
        Pw[(size_t)(r + 8 * lh) * 64 + t * 16 + lc] = (__bf16)p;
      }
#pragma unroll
    for (int r = 0; r < 8; ++r) {
#pragma unroll
      for (int m = 1; m < 16; m <<= 1) rsum[r] += __shfl_xor(rsum[r], m, 16);
      rl[r] = rl[r] * fac[r] + rsum[r];
    }
    // rescale accumulators
#pragma unroll
    for (int ct = 0; ct < 16; ++ct)
#pragma unroll
      for (int r = 0; r < 8; ++r) O[ct][r] *= fac[r];

    __builtin_amdgcn_wave_barrier();   // order P stores before fragment reads

    // O += P(16x64) x V(64x256) ; B operand from Vt[c][m] (K=m contiguous)
#pragma unroll
    for (int kk = 0; kk < 2; ++kk) {
      v16bf a = load_frag_a(Pw + kk * 32, 64);
#pragma unroll
      for (int ct = 0; ct < 16; ++ct) {
        v16bf bf = load_frag_b(Vc + (size_t)ct * 16 * 64 + kk * 32, 64);
        O[ct] = wmma_bf16(a, bf, O[ct]);
      }
    }
    __builtin_amdgcn_wave_barrier();

    // Next tile's TDM transfer must have landed before anyone reads it, and
    // everyone must be done with this tile before it is overwritten.
    wait_tiles();
  }

  // normalize, add residual f (f32 native layout), write refined tokens (bf16)
  float inv[8];
#pragma unroll
  for (int r = 0; r < 8; ++r) inv[r] = 1.0f / rl[r];
#pragma unroll
  for (int ct = 0; ct < 16; ++ct) {
    const int c = ct * 16 + lc;
    const float* fp = rb + (size_t)c * Nn + my_r0;   // 8 consecutive tokens
#pragma unroll
    for (int r = 0; r < 8; ++r) {
      float o = O[ct][r] * inv[r] + fp[r];
      ob[(size_t)(my_r0 + r) * Cc + c] = (__bf16)o;
    }
  }
}

// ---------------------------------------------------------------------------
// Output head: out[b][slot][m][n] = sum_k Wl1[m][k]*refined[n][k]
//                                 + sum_k Wl2[m][k]*f[n][k] + bl[m]
// grid (49, 4, B), block 128
// ---------------------------------------------------------------------------
__global__ void gemm_out_kernel(const __bf16* __restrict__ Wl1b,
                                const __bf16* __restrict__ Wl2b,
                                const __bf16* __restrict__ rtok,
                                const __bf16* __restrict__ ftok,
                                const float* __restrict__ bl,
                                float* __restrict__ out, int slot) {
  const int b = blockIdx.z;
  const int wave = threadIdx.x >> 5, lane = threadIdx.x & 31;
  const int lc = lane & 15, lh = lane >> 4;
  const int m0 = blockIdx.y * 64 + wave * 16;
  const int n0 = blockIdx.x * 64;
  const __bf16* rbp = rtok + (size_t)b * Nn * Cc;
  const __bf16* fbp = ftok + (size_t)b * Nn * Cc;

  v8f acc[4];
#pragma unroll
  for (int t = 0; t < 4; ++t)
#pragma unroll
    for (int r = 0; r < 8; ++r) acc[t][r] = bl[m0 + r + 8 * lh];

#pragma unroll
  for (int kc = 0; kc < 8; ++kc) {
    v16bf a1 = load_frag_a(Wl1b + (size_t)m0 * Cc + kc * 32, Cc);
    v16bf a2 = load_frag_a(Wl2b + (size_t)m0 * Cc + kc * 32, Cc);
#pragma unroll
    for (int t = 0; t < 4; ++t) {
      v16bf b1 = load_frag_b(rbp + (size_t)(n0 + t * 16) * Cc + kc * 32, Cc);
      acc[t] = wmma_bf16(a1, b1, acc[t]);
      v16bf b2 = load_frag_b(fbp + (size_t)(n0 + t * 16) * Cc + kc * 32, Cc);
      acc[t] = wmma_bf16(a2, b2, acc[t]);
    }
  }
  float* ob = out + (size_t)(b * 4 + slot) * Cc * Nn;
#pragma unroll
  for (int t = 0; t < 4; ++t)
#pragma unroll
    for (int r = 0; r < 8; ++r)
      ob[(size_t)(m0 + r + 8 * lh) * Nn + n0 + t * 16 + lc] = acc[t][r];
}

// ---------------------------------------------------------------------------
// Host launcher
// ---------------------------------------------------------------------------
extern "C" void kernel_launch(void* const* d_in, const int* in_sizes, int n_in,
                              void* d_out, int out_size, void* d_ws,
                              size_t ws_size, hipStream_t stream) {
  (void)in_sizes; (void)n_in; (void)out_size; (void)ws_size;
  const float* feat[4] = {(const float*)d_in[0], (const float*)d_in[1],
                          (const float*)d_in[2], (const float*)d_in[3]};
  const float* mask = (const float*)d_in[7];
  const float* Wv = (const float*)d_in[8];
  const float* bv = (const float*)d_in[9];
  const float* Wl = (const float*)d_in[10];
  const float* bl = (const float*)d_in[11];
  float* out = (float*)d_out;

  // workspace layout (all bf16): qA | qB | ftok | vT | Wvb | Wl1 | Wl2
  const size_t TOK = (size_t)Bb * Nn * Cc;       // elements per token buffer
  __bf16* qA = (__bf16*)d_ws;
  __bf16* qB = qA + TOK;
  __bf16* ftok = qB + TOK;
  __bf16* vT = ftok + TOK;
  __bf16* Wvb = vT + TOK;
  __bf16* Wl1 = Wvb + (size_t)Cc * Cc;
  __bf16* Wl2 = Wl1 + (size_t)Cc * Cc;
  // total ≈ 4*6.4MB + 0.4MB ≈ 26MB of workspace

  conv_weights_kernel<<<(Cc * Cc + 255) / 256, 256, 0, stream>>>(Wv, Wl, Wvb,
                                                                 Wl1, Wl2);
  copy_feat0_kernel<<<(Bb * Cc * Nn + 255) / 256, 256, 0, stream>>>(feat[0],
                                                                    out);
  transpose_tok_kernel<<<dim3(Nn / 32, Cc / 32, Bb), 256, 0, stream>>>(feat[0],
                                                                       qA);

  const size_t attn_lds =
      (size_t)(64 * Cc + 2 * 64 * Cc + 2 * Cc * 64 + 4 * 16 * 64) *
      sizeof(__bf16);   // 168 KB

  __bf16 *q = qA, *qn = qB;
  for (int s = 0; s < 3; ++s) {
    const float* fk = feat[s + 1];
    transpose_tok_kernel<<<dim3(Nn / 32, Cc / 32, Bb), 256, 0, stream>>>(fk,
                                                                         ftok);
    gemm_vlinear_kernel<<<dim3(Nn / 64, Cc / 64, Bb), 128, 0, stream>>>(
        Wvb, ftok, bv, vT);
    attn_kernel<<<dim3(Nn / 64, Bb), 128, attn_lds, stream>>>(q, ftok, vT,
                                                              mask, fk, qn);
    gemm_out_kernel<<<dim3(Nn / 64, Cc / 64, Bb), 128, 0, stream>>>(
        Wl1, Wl2, qn, ftok, bl, out, s + 1);
    __bf16* tmp = q; q = qn; qn = tmp;
  }
}
